// Attention_77653008712365
// MI455X (gfx1250) — matrix operand order repaired and checked by reading, not compiled, not run on hardware
//
#include <hip/hip_runtime.h>

// Problem constants (from reference): BS=2, S=2048, D=1024, H=16, DH=64
#define BSZ    2
#define SEQ    2048
#define DMODEL 1024
#define NH     16
#define DHD    64
#define NQKV   3072            // H * 3*DH
#define MROWS  4096            // BS * S

typedef __attribute__((ext_vector_type(16))) __bf16         v16bf;
typedef __attribute__((ext_vector_type(8)))  float          v8f;
typedef __attribute__((ext_vector_type(8)))  unsigned short v8us;
typedef __attribute__((ext_vector_type(16))) unsigned short v16us;

__device__ __forceinline__ unsigned short f2bf(float f) {
    unsigned int u = __float_as_uint(f);
    u += 0x7FFFu + ((u >> 16) & 1u);        // round-to-nearest-even
    return (unsigned short)(u >> 16);
}

// A-fragment (16x32 bf16): lane<16: e0..15=K0..15 ; lane>=16: K16..31
__device__ __forceinline__ v16bf load_frag_a(const unsigned short* p0,
                                             const unsigned short* p1) {
    v8us lo = *(const v8us*)p0;
    v8us hi = *(const v8us*)p1;
    v16us t;
#pragma unroll
    for (int i = 0; i < 8; ++i) { t[i] = lo[i]; t[i + 8] = hi[i]; }
    return __builtin_bit_cast(v16bf, t);
}

// B-fragment (32x16 bf16): lane<16: e0..15=K0..15 ; lane>=16: K16..31
__device__ __forceinline__ v16bf load_frag_b(const unsigned short* p) {
    v8us lo = *(const v8us*)p;
    v8us hi = *(const v8us*)(p + 8);
    v16us t;
#pragma unroll
    for (int i = 0; i < 8; ++i) { t[i] = lo[i]; t[i + 8] = hi[i]; }
    return __builtin_bit_cast(v16bf, t);
}

__device__ __forceinline__ v8f wmma_bf16(v16bf a, v16bf b, v8f c) {
    return __builtin_amdgcn_wmma_f32_16x16x32_bf16(false, a, false, b,
                                                   (short)0, c, false, false);
}

// ---- CDNA5 async global->LDS copy (ASYNCcnt-tracked, no VGPR round-trip) ----
// GVS addressing: mem = SGPR64 base + VGPR32 offset; LDS dest = VGPR32 byte addr
__device__ __forceinline__ void async_ld_b128(unsigned lds_byte_off,
                                              const void* sgpr_base,
                                              unsigned gbyte_off) {
    asm volatile("global_load_async_to_lds_b128 %0, %1, %2"
                 :
                 : "v"(lds_byte_off), "v"(gbyte_off), "s"(sgpr_base)
                 : "memory");
}
#define WAIT_ASYNC(N) asm volatile("s_wait_asynccnt " #N ::: "memory")

// ---------------- pack / convert kernels ----------------
__global__ void pack_x_kernel(const float* __restrict__ x,
                              unsigned short* __restrict__ xb) {
    int i = blockIdx.x * blockDim.x + threadIdx.x;   // MROWS*DMODEL
    xb[i] = f2bf(x[i]);
}

// wqkvb[n][d] = w_qkv[d][n], n = h*192+f
__global__ void pack_wqkv_kernel(const float* __restrict__ w,
                                 unsigned short* __restrict__ wb) {
    int i = blockIdx.x * blockDim.x + threadIdx.x;   // NQKV*DMODEL
    int n = i >> 10, d = i & (DMODEL - 1);
    wb[i] = f2bf(w[d * NQKV + n]);
}

// woutb[o][hd] = w_out[hd][o]
__global__ void pack_wout_kernel(const float* __restrict__ w,
                                 unsigned short* __restrict__ wb) {
    int i = blockIdx.x * blockDim.x + threadIdx.x;   // DMODEL*DMODEL
    int o = i >> 10, hd = i & (DMODEL - 1);
    wb[i] = f2bf(w[hd * DMODEL + o]);
}

// -------- shared 128x128 block-tile mainloop (double-buffered async->LDS) ----
// smem layout (ushorts): buf0:A[128*32] B[128*32] buf1:A B  (32 KB total)
// wave (8/block): mw = wave>>1 in 0..3 (M 32 each), nw = wave&1 (N 64 each)
__device__ __forceinline__ void
gemm128_mainloop(const unsigned short* __restrict__ Ag,   // [M x K] row-major
                 const unsigned short* __restrict__ Bg,   // [N x K] row-major
                 int m0b, int n0b, int K,
                 unsigned short* smem, v8f (&acc)[2][4]) {
    const int tid  = threadIdx.x;
    const int lane = tid & 31, wave = tid >> 5;
    const int r = lane & 15, hi = lane >> 4;
    const int mw = wave >> 1, nw = wave & 1;
    const unsigned sbase = (unsigned)(size_t)smem;       // LDS byte offset (flat low bits)
    const int NK = K / 32;

    // issue one 128x32 ushort tile pair: 512 chunks of 16B each, 2 per thread
    auto issue = [&](int kk, int buf) {
        const unsigned abase = sbase + (unsigned)buf * 16384u;
        const unsigned bbase = abase + 8192u;
        const unsigned kus = (unsigned)kk * 32u;
#pragma unroll
        for (int t = 0; t < 2; ++t) {
            unsigned c = (unsigned)tid + (unsigned)t * 256u;
            unsigned row = c >> 2, seg = (c & 3) * 8u;
            unsigned loff = (row * 32u + seg) * 2u;
            async_ld_b128(abase + loff, Ag,
                          (((unsigned)m0b + row) * (unsigned)K + kus + seg) * 2u);
            async_ld_b128(bbase + loff, Bg,
                          (((unsigned)n0b + row) * (unsigned)K + kus + seg) * 2u);
        }
    };

    issue(0, 0);
    for (int kk = 0; kk < NK; ++kk) {
        const int buf = kk & 1;
        if (kk + 1 < NK) { issue(kk + 1, buf ^ 1); WAIT_ASYNC(4); }
        else             { WAIT_ASYNC(0); }
        __syncthreads();

        const unsigned short* At = smem + buf * 8192;
        const unsigned short* Bt = At + 4096;
        const unsigned short* aL = At + (mw * 32 + r) * 32;
        const unsigned short* aH = At + (mw * 32 + 16 + r) * 32;
        v16bf a0 = load_frag_a(aL + hi * 16, aL + 8 + hi * 16);
        v16bf a1 = load_frag_a(aH + hi * 16, aH + 8 + hi * 16);
#pragma unroll
        for (int t = 0; t < 4; ++t) {
            v16bf b = load_frag_b(Bt + (nw * 64 + 16 * t + r) * 32 + hi * 16);
            acc[0][t] = wmma_bf16(a0, b, acc[0][t]);
            acc[1][t] = wmma_bf16(a1, b, acc[1][t]);
        }
        __syncthreads();
    }
}

// ---------------- QKV projection GEMM ----------------
// [4096 x 1024] x [3072 x 1024]^T ; 128x128/block, 32x64/wave
__global__ void __launch_bounds__(256)
qkv_gemm_kernel(const unsigned short* __restrict__ xb,
                const unsigned short* __restrict__ wb,
                const float* __restrict__ bqkv,
                unsigned short* __restrict__ qb,
                unsigned short* __restrict__ kbuf,
                unsigned short* __restrict__ vtb) {
    __shared__ unsigned short smem[4 * 4096];
    const int lane = threadIdx.x & 31, wave = threadIdx.x >> 5;
    const int r = lane & 15, hi = lane >> 4;
    const int mw = wave >> 1, nw = wave & 1;
    const int m0b = (blockIdx.x / (NQKV / 128)) * 128;   // 32 x 24 blocks
    const int n0b = (blockIdx.x % (NQKV / 128)) * 128;

    v8f acc[2][4] = {};
    gemm128_mainloop(xb, wb, m0b, n0b, DMODEL, smem, acc);

    // epilogue: split into Q (scaled), K, V^T (all bf16)
#pragma unroll
    for (int t = 0; t < 4; ++t) {
        const int n = n0b + nw * 64 + 16 * t + r;
        const int h = n / 192;
        const int f = n % 192;
        const float bias = bqkv[n];
#pragma unroll
        for (int i = 0; i < 2; ++i) {
#pragma unroll
            for (int j = 0; j < 8; ++j) {
                const int m  = m0b + mw * 32 + i * 16 + hi * 8 + j;
                const int b_ = m >> 11;              // / SEQ
                const int s  = m & (SEQ - 1);
                const float v = acc[i][t][j] + bias;
                if (f < 64) {
                    qb[((b_ * NH + h) * SEQ + s) * DHD + f] = f2bf(v * 0.125f);
                } else if (f < 128) {
                    kbuf[((b_ * NH + h) * SEQ + s) * DHD + (f - 64)] = f2bf(v);
                } else {
                    vtb[((b_ * NH + h) * DHD + (f - 128)) * SEQ + s] = f2bf(v);
                }
            }
        }
    }
}

// ---------------- flash attention (causal) ----------------
// workgroup = 128 consecutive q rows of one (b,h); 8 waves x 16 rows.
// K/V key-blocks (32 keys) staged once per workgroup via async->LDS, double buffered.
// smem (ushorts): buf0 K[32x64] V[64x32] | buf1 K V | P-scratch 8x(16x40)
__global__ void __launch_bounds__(256)
attn_kernel(const unsigned short* __restrict__ qb,
            const unsigned short* __restrict__ kbuf,
            const unsigned short* __restrict__ vtb,
            unsigned short* __restrict__ attnb) {
    __shared__ unsigned short smem[2 * 4096 + 8 * 640];  // 16 KB KV + 10 KB P
    const int tid  = threadIdx.x;
    const int lane = tid & 31, wave = tid >> 5;
    const int r = lane & 15, hi = lane >> 4;

    const int qblocks = SEQ / 128;                       // 16
    const int bh  = blockIdx.x / qblocks;                // 0..31 (b*NH+h)
    const int qsb = (blockIdx.x % qblocks) * 128;        // block q start
    const int qs  = qsb + wave * 16;                     // wave q start

    const unsigned short* qbase = qb   + bh * SEQ * DHD;
    const unsigned short* kbase = kbuf + bh * SEQ * DHD; // uniform SGPR base
    const unsigned short* vbase = vtb  + bh * DHD * SEQ; // uniform SGPR base

    unsigned short* pw = smem + 8192 + wave * 640;
    const unsigned sbase = (unsigned)(size_t)smem;

    // stage one 32-key K/V block into buffer buf (2 async b128 per thread)
    auto stage = [&](int kb, int buf) {
        const unsigned kdst = sbase + (unsigned)buf * 8192u;   // K tile 32x64
        const unsigned vdst = kdst + 4096u;                    // V tile 64x32
        {   // K: row = tid>>3 (0..31), seg = (tid&7)*8
            unsigned row = (unsigned)tid >> 3, seg = ((unsigned)tid & 7u) * 8u;
            async_ld_b128(kdst + (row * 64u + seg) * 2u, kbase,
                          (((unsigned)kb + row) * 64u + seg) * 2u);
        }
        {   // V^T: row(d) = tid>>2 (0..63), seg = (tid&3)*8
            unsigned row = (unsigned)tid >> 2, seg = ((unsigned)tid & 3u) * 8u;
            async_ld_b128(vdst + (row * 32u + seg) * 2u, vbase,
                          (row * (unsigned)SEQ + (unsigned)kb + seg) * 2u);
        }
    };

    // preload this wave's Q A-fragments (d = 0..31, 32..63)
    const unsigned short* qrow = qbase + (qs + r) * DHD;
    const v16bf qf0 = load_frag_a(qrow + hi * 16,      qrow + 8 + hi * 16);
    const v16bf qf1 = load_frag_a(qrow + 32 + hi * 16, qrow + 40 + hi * 16);

    float mrow[8], lrow[8];
    v8f acc0 = {}, acc1 = {}, acc2 = {}, acc3 = {};
#pragma unroll
    for (int j = 0; j < 8; ++j) { mrow[j] = -3.0e38f; lrow[j] = 0.0f; }

    const int kend_blk = qsb + 128;   // uniform across workgroup
    const int kend_w   = qs + 16;     // this wave's causal bound
    stage(0, 0);
    for (int kb = 0; kb < kend_blk; kb += 32) {
        const int buf = (kb >> 5) & 1;
        if (kb + 32 < kend_blk) { stage(kb + 32, buf ^ 1); WAIT_ASYNC(2); }
        else                    { WAIT_ASYNC(0); }
        __syncthreads();

        if (kb < kend_w) {            // wave-uniform guard: EXEC all-1 inside
            const unsigned short* Kt = smem + buf * 4096;
            const unsigned short* Vt = Kt + 2048;

            // ---- scores S = Q K^T for 32 keys (two 16-col strips) ----
            const unsigned short* k0row = Kt + r * DHD;
            const unsigned short* k1row = Kt + (16 + r) * DHD;
            v8f s0 = {}, s1 = {};
            s0 = wmma_bf16(qf0, load_frag_b(k0row + hi * 16), s0);
            s0 = wmma_bf16(qf1, load_frag_b(k0row + 32 + hi * 16), s0);
            s1 = wmma_bf16(qf0, load_frag_b(k1row + hi * 16), s1);
            s1 = wmma_bf16(qf1, load_frag_b(k1row + 32 + hi * 16), s1);

            // ---- causal mask + online softmax ----
            float p0[8], p1[8];
#pragma unroll
            for (int j = 0; j < 8; ++j) {
                const int q = qs + hi * 8 + j;
                float v0 = (kb + r      <= q) ? s0[j] : -3.0e38f;
                float v1 = (kb + 16 + r <= q) ? s1[j] : -3.0e38f;
                float mx = fmaxf(v0, v1);
#pragma unroll
                for (int msk = 1; msk <= 8; msk <<= 1)
                    mx = fmaxf(mx, __shfl_xor(mx, msk, 32));
                const float mnew  = fmaxf(mrow[j], mx);
                const float alpha = __expf(mrow[j] - mnew);
                mrow[j] = mnew;
                v0 = __expf(v0 - mnew);
                v1 = __expf(v1 - mnew);
                float sum = v0 + v1;
#pragma unroll
                for (int msk = 1; msk <= 8; msk <<= 1)
                    sum += __shfl_xor(sum, msk, 32);
                lrow[j] = lrow[j] * alpha + sum;
                p0[j] = v0; p1[j] = v1;
                acc0[j] *= alpha; acc1[j] *= alpha;
                acc2[j] *= alpha; acc3[j] *= alpha;
            }

            // ---- C-layout P -> LDS -> A-fragment layout ----
#pragma unroll
            for (int j = 0; j < 8; ++j) {
                pw[(hi * 8 + j) * 40 + r]      = f2bf(p0[j]);
                pw[(hi * 8 + j) * 40 + 16 + r] = f2bf(p1[j]);
            }
            asm volatile("s_wait_dscnt 0" ::: "memory");
            const unsigned short* prow = pw + r * 40;
            const v16bf pf = load_frag_a(prow + hi * 16, prow + 8 + hi * 16);

            // ---- O += P V (4 d-tiles of 16) ----
            acc0 = wmma_bf16(pf, load_frag_b(Vt + (r)      * 32 + hi * 16), acc0);
            acc1 = wmma_bf16(pf, load_frag_b(Vt + (16 + r) * 32 + hi * 16), acc1);
            acc2 = wmma_bf16(pf, load_frag_b(Vt + (32 + r) * 32 + hi * 16), acc2);
            acc3 = wmma_bf16(pf, load_frag_b(Vt + (48 + r) * 32 + hi * 16), acc3);
        }
        __syncthreads();
    }

    // ---- normalize and write attn [B,S,H*DH] bf16 ----
    const int b_ = bh / NH, h = bh % NH;
#pragma unroll
    for (int j = 0; j < 8; ++j) {
        const float inv = 1.0f / lrow[j];
        unsigned short* orow =
            attnb + (size_t)(b_ * SEQ + qs + hi * 8 + j) * DMODEL + h * DHD;
        orow[r]      = f2bf(acc0[j] * inv);
        orow[16 + r] = f2bf(acc1[j] * inv);
        orow[32 + r] = f2bf(acc2[j] * inv);
        orow[48 + r] = f2bf(acc3[j] * inv);
    }
}

// ---------------- output projection GEMM ----------------
// [4096 x 1024] x [1024 x 1024]^T + bias -> f32 out
__global__ void __launch_bounds__(256)
out_gemm_kernel(const unsigned short* __restrict__ ab,
                const unsigned short* __restrict__ wb,
                const float* __restrict__ bout,
                float* __restrict__ out) {
    __shared__ unsigned short smem[4 * 4096];
    const int lane = threadIdx.x & 31, wave = threadIdx.x >> 5;
    const int r = lane & 15, hi = lane >> 4;
    const int mw = wave >> 1, nw = wave & 1;
    const int m0b = (blockIdx.x / (DMODEL / 128)) * 128; // 32 x 8 blocks
    const int n0b = (blockIdx.x % (DMODEL / 128)) * 128;

    v8f acc[2][4] = {};
    gemm128_mainloop(ab, wb, m0b, n0b, DMODEL, smem, acc);

#pragma unroll
    for (int t = 0; t < 4; ++t) {
        const int n = n0b + nw * 64 + 16 * t + r;
        const float bias = bout[n];
#pragma unroll
        for (int i = 0; i < 2; ++i) {
#pragma unroll
            for (int j = 0; j < 8; ++j) {
                const int m = m0b + mw * 32 + i * 16 + hi * 8 + j;
                out[(size_t)m * DMODEL + n] = acc[i][t][j] + bias;
            }
        }
    }
}

// ---------------- launcher ----------------
extern "C" void kernel_launch(void* const* d_in, const int* in_sizes, int n_in,
                              void* d_out, int out_size, void* d_ws, size_t ws_size,
                              hipStream_t stream) {
    (void)in_sizes; (void)n_in; (void)out_size; (void)ws_size;
    const float* x     = (const float*)d_in[0];
    const float* w_qkv = (const float*)d_in[1];
    const float* b_qkv = (const float*)d_in[2];
    const float* w_out = (const float*)d_in[3];
    const float* b_out = (const float*)d_in[4];
    float* out = (float*)d_out;

    char* ws = (char*)d_ws;
    const size_t MB = 1024 * 1024;
    unsigned short* xb    = (unsigned short*)(ws);            //  8 MB
    unsigned short* wqkvb = (unsigned short*)(ws +  8 * MB);  //  6 MB
    unsigned short* woutb = (unsigned short*)(ws + 14 * MB);  //  2 MB
    unsigned short* qb    = (unsigned short*)(ws + 16 * MB);  //  8 MB
    unsigned short* kb    = (unsigned short*)(ws + 24 * MB);  //  8 MB
    unsigned short* vtb   = (unsigned short*)(ws + 32 * MB);  //  8 MB
    unsigned short* attnb = (unsigned short*)(ws + 40 * MB);  //  8 MB (48 MB total)

    pack_x_kernel   <<<(MROWS * DMODEL) / 256, 256, 0, stream>>>(x, xb);
    pack_wqkv_kernel<<<(NQKV * DMODEL) / 256, 256, 0, stream>>>(w_qkv, wqkvb);
    pack_wout_kernel<<<(DMODEL * DMODEL) / 256, 256, 0, stream>>>(w_out, woutb);

    qkv_gemm_kernel<<<(MROWS / 128) * (NQKV / 128), 256, 0, stream>>>(
        xb, wqkvb, b_qkv, qb, kb, vtb);
    attn_kernel<<<BSZ * NH * (SEQ / 128), 256, 0, stream>>>(qb, kb, vtb, attnb);
    out_gemm_kernel<<<(MROWS / 128) * (DMODEL / 128), 256, 0, stream>>>(
        attnb, woutb, b_out, out);
}
